// MMDiTAttention_68023692034118
// MI455X (gfx1250) — compile-verified, matrix operand order checked by reading, not run on hardware
//
#include <hip/hip_runtime.h>

// ---------------------------------------------------------------------------
// MMDiT joint attention for MI455X (gfx1250, wave32, WMMA).
// Pipeline: [gemm qkv_in] [gemm qkv_ctx] -> [rmsnorm q/k + f16 pack]
//           -> [flash attention, f16 WMMA + async global->LDS] -> [gemm out_*]
// All matrix math on v_wmma_f32_16x16x32_f16 (fp32 inputs converted to f16
// in the LDS staging step; f32 accumulate).
// ---------------------------------------------------------------------------

typedef __attribute__((ext_vector_type(16))) _Float16 v16h;
typedef __attribute__((ext_vector_type(8)))  float    v8f;

#define WMMA_F16(a, b, c) \
    __builtin_amdgcn_wmma_f32_16x16x32_f16(false, (a), false, (b), (short)0, (c), false, false)

// Load a 16x32 f16 fragment (A layout; B 32x16 uses the mirrored layout with
// "row" = column index) from LDS, ISA 7.12.2:
//   lane L: M = L%16 ; K-base = (L>=16 ? 8 : 0); VGPR0..3 hold K=kb..kb+7,
//   VGPR4..7 hold K=kb+16..kb+23  -> two contiguous 128-bit loads per lane.
__device__ __forceinline__ v16h ldfrag(const _Float16* base, int row0, int stride, int k)
{
    const int lane = threadIdx.x & 31;
    const int r    = lane & 15;
    const int kb   = (lane >> 4) << 3;
    const _Float16* p = base + (size_t)(row0 + r) * (size_t)stride + (size_t)(k + kb);
    v16h f;
    ((uint4*)&f)[0] = *(const uint4*)(p);
    ((uint4*)&f)[1] = *(const uint4*)(p + 16);
    return f;
}

// Async global -> LDS copy, 16B per lane per issue (GLOBAL_LOAD_ASYNC_TO_LDS_B128,
// GV mode, ASYNCcnt tracked).  On gfx1250 the low 32 bits of a flat pointer to
// LDS are the wave-relative LDS byte address (ISA 10.2 aperture rules), which is
// exactly what the VDST operand wants.
__device__ __forceinline__ void async_copy_b128(void* lds_dst, const void* gsrc)
{
    asm volatile("global_load_async_to_lds_b128 %0, %1, off"
                 :
                 : "v"((unsigned)(uintptr_t)lds_dst),
                   "v"((unsigned long long)(uintptr_t)gsrc)
                 : "memory");
}
__device__ __forceinline__ void async_wait0()
{
    asm volatile("s_wait_asynccnt 0x0" ::: "memory");
}

// ---------------------------------------------------------------------------
// Generic tiled GEMM:  C[b, c_row0+r, :] = A[b, a_row0+r, :] @ W + bias
// A: fp32, [batch][seg rows][lda];  W: fp32 [K x N] row-major;  C: fp32.
// BM=128 x BN=128 x BK=32 tile, 256 threads = 8 waves (4x2), each wave 32x64.
// seg is always a multiple of BM, so a tile never crosses a batch boundary:
// batch decomposition is hoisted to one division per workgroup.
// ---------------------------------------------------------------------------
#define BM 128
#define BN 128
#define BK 32

__global__ void __launch_bounds__(256)
wmma_gemm_bias(const float* __restrict__ A, int lda, long a_bs, int a_row0, int seg,
               const float* __restrict__ W, const float* __restrict__ bias,
               float* __restrict__ C, int ldc, long c_bs, int c_row0,
               int M, int N, int K)
{
    __shared__ __align__(16) _Float16 As[BM * BK];   // [m][k]
    __shared__ __align__(16) _Float16 Bs[BN * BK];   // [n][k] (transposed)

    const int tid = threadIdx.x;
    const int m0  = blockIdx.y * BM;
    const int n0  = blockIdx.x * BN;
    const int w   = tid >> 5;
    const int wm  = w >> 1;     // 0..3 -> M offset wm*32
    const int wn  = w & 1;      // 0..1 -> N offset wn*64
    const int lane = tid & 31;

    // Hoisted batch decomposition (tile fully inside one batch).
    const int  b    = m0 / seg;
    const int  rrow = m0 - b * seg;
    const float* Abase = A + (long)b * a_bs + (long)(a_row0 + rrow) * lda;
    float*       Cbase = C + (long)b * c_bs + (long)(c_row0 + rrow) * ldc;

    const v8f vz = {0.f, 0.f, 0.f, 0.f, 0.f, 0.f, 0.f, 0.f};
    v8f acc[2][4];
    for (int i = 0; i < 2; ++i)
        for (int j = 0; j < 4; ++j) acc[i][j] = vz;

    for (int k0 = 0; k0 < K; k0 += BK) {
        __syncthreads();
        // A tile: 128x32 fp32 -> f16 LDS (row-major, stride BK)
        for (int i = tid; i < BM * BK / 4; i += 256) {
            const int r  = i >> 3;
            const int kq = (i & 7) << 2;
            const float* src = Abase + (long)r * lda + k0 + kq;
            const float4 v = *(const float4*)src;
            _Float16* dst = As + r * BK + kq;
            dst[0] = (_Float16)v.x; dst[1] = (_Float16)v.y;
            dst[2] = (_Float16)v.z; dst[3] = (_Float16)v.w;
            if (k0 + BK < K) __builtin_prefetch(src + BK, 0, 1);
        }
        // W tile: 32x128 fp32 -> f16 LDS transposed [n][k]
        for (int i = tid; i < BK * BN / 4; i += 256) {
            const int k  = i >> 5;
            const int nq = (i & 31) << 2;
            const float* src = W + (long)(k0 + k) * N + n0 + nq;
            const float4 v = *(const float4*)src;
            Bs[(nq + 0) * BK + k] = (_Float16)v.x;
            Bs[(nq + 1) * BK + k] = (_Float16)v.y;
            Bs[(nq + 2) * BK + k] = (_Float16)v.z;
            Bs[(nq + 3) * BK + k] = (_Float16)v.w;
            if (k0 + BK < K) __builtin_prefetch(src + (long)BK * N, 0, 1);
        }
        __syncthreads();

        const v16h a0 = ldfrag(As, wm * 32,      BK, 0);
        const v16h a1 = ldfrag(As, wm * 32 + 16, BK, 0);
        for (int nt = 0; nt < 4; ++nt) {
            const v16h bfrag = ldfrag(Bs, wn * 64 + nt * 16, BK, 0);
            acc[0][nt] = WMMA_F16(a0, bfrag, acc[0][nt]);
            acc[1][nt] = WMMA_F16(a1, bfrag, acc[1][nt]);
        }
    }

    // Epilogue: C/D layout -> lane: N = lane%16, M = v + 8*(lane>=16)
    const int n  = lane & 15;
    const int hi = lane >> 4;
    for (int mt = 0; mt < 2; ++mt)
        for (int nt = 0; nt < 4; ++nt)
            for (int v = 0; v < 8; ++v) {
                const int rl  = wm * 32 + mt * 16 + hi * 8 + v;
                const int col = n0 + wn * 64 + nt * 16 + n;
                Cbase[(long)rl * ldc + col] = acc[mt][nt][v] + bias[col];
            }
}

// ---------------------------------------------------------------------------
// RMSNorm over full DIM=1024 for q and k chunks of unified qkv[B,4608,3072],
// select ctx/input scale per row, emit f16 Q/K/V.
// ---------------------------------------------------------------------------
__global__ void __launch_bounds__(256)
rmsnorm_pack(const float* __restrict__ qkv,
             const float* __restrict__ qs_in, const float* __restrict__ ks_in,
             const float* __restrict__ qs_cx, const float* __restrict__ ks_cx,
             _Float16* __restrict__ Qh, _Float16* __restrict__ Kh,
             _Float16* __restrict__ Vh)
{
    __shared__ float red[2][256];
    const int  row    = blockIdx.x;                 // 0 .. B*4608-1
    const int  tid    = threadIdx.x;
    const bool is_ctx = (row % 4608) < 512;
    const float* qs = is_ctx ? qs_cx : qs_in;
    const float* ks = is_ctx ? ks_cx : ks_in;
    const float* base = qkv + (long)row * 3072;

    float qv[4], kv[4], sq = 0.f, sk = 0.f;
    for (int j = 0; j < 4; ++j) {
        const int d = tid + j * 256;
        qv[j] = base[d];
        kv[j] = base[1024 + d];
        sq += qv[j] * qv[j];
        sk += kv[j] * kv[j];
    }
    red[0][tid] = sq; red[1][tid] = sk;
    __syncthreads();
    for (int s = 128; s > 0; s >>= 1) {
        if (tid < s) { red[0][tid] += red[0][tid + s]; red[1][tid] += red[1][tid + s]; }
        __syncthreads();
    }
    const float rq = rsqrtf(red[0][0] * (1.f / 1024.f) + 1e-6f);
    const float rk = rsqrtf(red[1][0] * (1.f / 1024.f) + 1e-6f);
    for (int j = 0; j < 4; ++j) {
        const int  d = tid + j * 256;
        const long o = (long)row * 1024 + d;
        Qh[o] = (_Float16)(qv[j] * rq * qs[d]);
        Kh[o] = (_Float16)(kv[j] * rk * ks[d]);
        Vh[o] = (_Float16)base[2048 + d];
    }
}

// ---------------------------------------------------------------------------
// Flash attention: 32 query rows / WG, KV step 32, D=1024, S=4608.
// 256 threads = 8 waves. Q and K tiles staged with GLOBAL_LOAD_ASYNC_TO_LDS.
// S=Q.K^T split 4 tiles x 2 K-halves across waves (partials combined in LDS);
// 32 lanes run online softmax; all waves hold a 32x128 f32 accumulator slab
// and do PV via WMMA.
// Dynamic LDS: Qs 64KB | KVs 64KB (K tile, then V^T tile) | Sp 8KB | Pm 2KB
//              | alpha 128B | linv 128B  => 141568 B.
// ---------------------------------------------------------------------------
#define QT   32
#define KT   32
#define ADIM 1024
#define STOT 4608

__global__ void __launch_bounds__(256)
flash_attn(const _Float16* __restrict__ Qh, const _Float16* __restrict__ Kh,
           const _Float16* __restrict__ Vh, float* __restrict__ O)
{
    extern __shared__ __align__(16) char smem[];
    _Float16* Qs     = (_Float16*)(smem);                       // [32][1024]
    _Float16* KVs    = (_Float16*)(smem + 65536);               // K:[32][1024] / V^T:[1024][32]
    float*    Sp     = (float*)  (smem + 131072);               // [8][16][16]
    _Float16* Pm     = (_Float16*)(smem + 131072 + 8192);       // [32][32]
    float*    alphaS = (float*)  (smem + 131072 + 8192 + 2048); // [32]
    float*    linvS  = alphaS + 32;                             // [32]

    const int tid   = threadIdx.x;
    const int lane  = tid & 31;
    const int w     = tid >> 5;
    const int batch = blockIdx.y;
    const int q0    = blockIdx.x * QT;

    // Stage Q tile (f16) into LDS asynchronously.
    {
        const uint4* src = (const uint4*)(Qh + ((long)batch * STOT + q0) * ADIM);
        uint4*       dst = (uint4*)Qs;
        for (int i = tid; i < QT * ADIM / 8; i += 256)
            async_copy_b128(dst + i, src + i);
        async_wait0();
    }

    const v8f vz = {0.f, 0.f, 0.f, 0.f, 0.f, 0.f, 0.f, 0.f};
    v8f acc[2][8];
    for (int i = 0; i < 2; ++i)
        for (int j = 0; j < 8; ++j) acc[i][j] = vz;
    float m_i = -3.0e38f, l_i = 0.f;

    const int t  = w & 3;        // S tile: mrow=(t>>1)*16, ncol=(t&1)*16
    const int kh = w >> 2;       // K half: 0 -> k in [0,512), 1 -> [512,1024)

    for (int kv0 = 0; kv0 < STOT; kv0 += KT) {
        __syncthreads();                                  // KVs reuse fence (+ Q visible)
        // K tile -> LDS (row-major [kvrow][d]) asynchronously
        {
            const uint4* src = (const uint4*)(Kh + ((long)batch * STOT + kv0) * ADIM);
            uint4*       dst = (uint4*)KVs;
            for (int i = tid; i < KT * ADIM / 8; i += 256)
                async_copy_b128(dst + i, src + i);
            async_wait0();
        }
        __syncthreads();

        // Partial S tile per wave: 16 WMMA k-steps over one 512-wide K half.
        {
            v8f s = vz;
            const int mrow = (t >> 1) * 16;
            const int ncol = (t & 1) * 16;
            for (int ks = 0; ks < 16; ++ks) {
                const int k = kh * 512 + ks * 32;
                const v16h a = ldfrag(Qs,  mrow, ADIM, k);   // Q rows
                const v16h b = ldfrag(KVs, ncol, ADIM, k);   // K rows = columns of K^T
                s = WMMA_F16(a, b, s);
            }
            float* spw = Sp + w * 256;
            const int n = lane & 15, hi = lane >> 4;
            for (int v = 0; v < 8; ++v) spw[(hi * 8 + v) * 16 + n] = s[v];
        }
        __syncthreads();

        // Online softmax, one row per lane of wave 0.
        if (tid < 32) {
            const int r = tid;
            float sv[32], rowmax = -3.0e38f;
            for (int c = 0; c < 32; ++c) {
                const int tt = ((r >> 4) << 1) + (c >> 4);
                float x = Sp[tt * 256 + (r & 15) * 16 + (c & 15)] +
                          Sp[(tt + 4) * 256 + (r & 15) * 16 + (c & 15)];
                x *= 0.125f;                               // SCALE = 64^-0.5
                sv[c] = x;
                rowmax = fmaxf(rowmax, x);
            }
            const float m_new = fmaxf(m_i, rowmax);
            const float alpha = __expf(m_i - m_new);
            float ps = 0.f;
            for (int c = 0; c < 32; ++c) {
                const float p = __expf(sv[c] - m_new);
                ps += p;
                Pm[r * 32 + c] = (_Float16)p;
            }
            l_i = l_i * alpha + ps;
            m_i = m_new;
            alphaS[r] = alpha;
        }
        __syncthreads();

        // V tile -> LDS transposed: KVs[d][kv]  (overwrites K tile)
        {
            const _Float16* vsrc = Vh + ((long)batch * STOT + kv0) * ADIM;
            for (int i = tid; i < KT * ADIM / 2; i += 256) {
                const int kvr = i / (ADIM / 2);
                const int d   = (i - kvr * (ADIM / 2)) * 2;
                union { unsigned u; _Float16 h[2]; } uu;
                uu.u = *(const unsigned*)(vsrc + (long)kvr * ADIM + d);
                KVs[(d + 0) * KT + kvr] = uu.h[0];
                KVs[(d + 1) * KT + kvr] = uu.h[1];
            }
        }
        __syncthreads();

        // Rescale accumulators and accumulate P @ V (each wave: cols w*128..+127)
        {
            const int hi = lane >> 4;
            for (int mt = 0; mt < 2; ++mt)
                for (int v = 0; v < 8; ++v) {
                    const float a = alphaS[mt * 16 + hi * 8 + v];
                    for (int nt = 0; nt < 8; ++nt) acc[mt][nt][v] *= a;
                }
            for (int mt = 0; mt < 2; ++mt) {
                const v16h ap = ldfrag(Pm, mt * 16, 32, 0);
                for (int nt = 0; nt < 8; ++nt) {
                    const v16h bv = ldfrag(KVs, w * 128 + nt * 16, 32, 0);
                    acc[mt][nt] = WMMA_F16(ap, bv, acc[mt][nt]);
                }
            }
        }
    }

    __syncthreads();
    if (tid < 32) linvS[tid] = 1.f / l_i;
    __syncthreads();

    {
        const int n = lane & 15, hi = lane >> 4;
        for (int mt = 0; mt < 2; ++mt)
            for (int nt = 0; nt < 8; ++nt)
                for (int v = 0; v < 8; ++v) {
                    const int rl  = mt * 16 + hi * 8 + v;
                    const int col = w * 128 + nt * 16 + n;
                    O[((long)batch * STOT + q0 + rl) * ADIM + col] =
                        acc[mt][nt][v] * linvS[rl];
                }
    }
}

// ---------------------------------------------------------------------------
// Launch
// ---------------------------------------------------------------------------
extern "C" void kernel_launch(void* const* d_in, const int* in_sizes, int n_in,
                              void* d_out, int out_size, void* d_ws, size_t ws_size,
                              hipStream_t stream)
{
    const float* input      = (const float*)d_in[0];   // [4,4096,1024]
    const float* context    = (const float*)d_in[1];   // [4,512,768]
    const float* w_qkv_in   = (const float*)d_in[2];   // [1024,3072]
    const float* b_qkv_in   = (const float*)d_in[3];
    const float* w_qkv_cx   = (const float*)d_in[4];   // [768,3072]
    const float* b_qkv_cx   = (const float*)d_in[5];
    const float* q_scale_in = (const float*)d_in[6];
    const float* k_scale_in = (const float*)d_in[7];
    const float* q_scale_cx = (const float*)d_in[8];
    const float* k_scale_cx = (const float*)d_in[9];
    const float* w_out_in   = (const float*)d_in[10];  // [1024,1024]
    const float* b_out_in   = (const float*)d_in[11];
    const float* w_out_cx   = (const float*)d_in[12];  // [1024,768]
    const float* b_out_cx   = (const float*)d_in[13];
    float* out = (float*)d_out;   // input_out [4,4096,1024] ++ context_out [4,512,768]

    // Workspace layout
    char*  ws  = (char*)d_ws;
    float* qkv = (float*)ws;                                 // [4][4608][3072] f32
    size_t off = (size_t)4 * 4608 * 3072 * sizeof(float);
    _Float16* Qh = (_Float16*)(ws + off); off += (size_t)4 * 4608 * 1024 * 2;
    _Float16* Kh = (_Float16*)(ws + off); off += (size_t)4 * 4608 * 1024 * 2;
    _Float16* Vh = (_Float16*)(ws + off); off += (size_t)4 * 4608 * 1024 * 2;
    float* attn  = (float*)(ws + off);                        // [4][4608][1024] f32

    // 1) QKV projection, input tokens -> qkv rows [512..4608) per batch
    wmma_gemm_bias<<<dim3(3072 / BN, 16384 / BM), 256, 0, stream>>>(
        input, 1024, (long)4096 * 1024, 0, 4096,
        w_qkv_in, b_qkv_in,
        qkv, 3072, (long)4608 * 3072, 512,
        16384, 3072, 1024);

    // 2) QKV projection, context tokens -> qkv rows [0..512) per batch
    wmma_gemm_bias<<<dim3(3072 / BN, 2048 / BM), 256, 0, stream>>>(
        context, 768, (long)512 * 768, 0, 512,
        w_qkv_cx, b_qkv_cx,
        qkv, 3072, (long)4608 * 3072, 0,
        2048, 3072, 768);

    // 3) QK RMSNorm + f16 packing of Q/K/V
    rmsnorm_pack<<<4 * 4608, 256, 0, stream>>>(
        qkv, q_scale_in, k_scale_in, q_scale_cx, k_scale_cx, Qh, Kh, Vh);

    // 4) Flash attention over unified sequence
    flash_attn<<<dim3(STOT / QT, 4), 256, 141568, stream>>>(Qh, Kh, Vh, attn);

    // 5) Output projection, input tokens (attn rows [512..4608) per batch)
    wmma_gemm_bias<<<dim3(1024 / BN, 16384 / BM), 256, 0, stream>>>(
        attn, 1024, (long)4608 * 1024, 512, 4096,
        w_out_in, b_out_in,
        out, 1024, (long)4096 * 1024, 0,
        16384, 1024, 1024);

    // 6) Output projection, context tokens (attn rows [0..512) per batch)
    wmma_gemm_bias<<<dim3(768 / BN, 2048 / BM), 256, 0, stream>>>(
        attn, 1024, (long)4608 * 1024, 0, 512,
        w_out_cx, b_out_cx,
        out + (size_t)4 * 4096 * 1024, 768, (long)512 * 768, 0,
        2048, 768, 1024);
}
